// SelfAttention2D_850403524739
// MI455X (gfx1250) — matrix operand order repaired and checked by reading, not compiled, not run on hardware
//
#include <hip/hip_runtime.h>

typedef _Float16 f16;
typedef __attribute__((ext_vector_type(16))) _Float16 v16h;
typedef __attribute__((ext_vector_type(8)))  _Float16 v8h;
typedef __attribute__((ext_vector_type(2)))  _Float16 v2h;
typedef __attribute__((ext_vector_type(8)))  float    v8f;
typedef __attribute__((ext_vector_type(8)))  int      v8i;

#define CIN   256
#define CQK   32
#define NPIX  4096
#define NB    4
#define LOG2E 1.4426950408889634f

// ---------------------------------------------------------------------------
// Kernel 1: concatenate wq|wk|wv into Wcat (320x256) and biases into bcat(320)
// ---------------------------------------------------------------------------
__global__ void concat_w_kernel(const float* __restrict__ wq, const float* __restrict__ bq,
                                const float* __restrict__ wk, const float* __restrict__ bk,
                                const float* __restrict__ wv, const float* __restrict__ bv,
                                float* __restrict__ Wcat, float* __restrict__ bcat) {
    int i = blockIdx.x * 256 + threadIdx.x;
    if (i < 320 * 256) {
        int d = i >> 8, c = i & 255;
        float v = (d < 32) ? wq[d * 256 + c]
                : (d < 64) ? wk[(d - 32) * 256 + c]
                           : wv[(d - 64) * 256 + c];
        Wcat[i] = v;
    }
    if (i < 320) {
        bcat[i] = (i < 32) ? bq[i] : (i < 64) ? bk[i - 32] : bv[i - 64];
    }
}

// ---------------------------------------------------------------------------
// Kernel 2: projections. Per block: one batch, 64-pixel tile, all 320 outputs.
// x tile (64c x 64n) and W tile (320d x 64c) staged in LDS per 64-channel chunk.
// Q,K stored (b, n, 32) f16 row-major; V stored (b, c, N) j-contiguous.
// ---------------------------------------------------------------------------
__global__ __launch_bounds__(512) void proj_kernel(
    const float* __restrict__ x, const float* __restrict__ Wcat,
    const float* __restrict__ bcat,
    f16* __restrict__ Qb, f16* __restrict__ Kb, f16* __restrict__ Vb) {
    __shared__ float xlds[64][64];
    __shared__ float wlds[320][64];
    const int t  = threadIdx.x;          // 512 threads
    const int b  = blockIdx.x >> 6;      // 64 n-tiles per batch
    const int n0 = (blockIdx.x & 63) * 64;
    const int nl = t & 63;
    const int dg = t >> 6;               // 0..7, each owns 40 d's

    float accum[40];
#pragma unroll
    for (int i = 0; i < 40; ++i) accum[i] = 0.0f;

    for (int cc = 0; cc < 4; ++cc) {     // 4 chunks of 64 input channels
#pragma unroll
        for (int i = 0; i < 8; ++i) {
            int idx = t + i * 512;
            int c = idx >> 6, n = idx & 63;
            xlds[c][n] = x[((size_t)b * CIN + cc * 64 + c) * NPIX + n0 + n];
        }
#pragma unroll
        for (int i = 0; i < 40; ++i) {
            int idx = t + i * 512;
            int d = idx >> 6, c = idx & 63;
            wlds[d][c] = Wcat[(size_t)d * CIN + cc * 64 + c];
        }
        __syncthreads();
        for (int c = 0; c < 64; ++c) {
            float xv = xlds[c][nl];
#pragma unroll
            for (int dd = 0; dd < 40; ++dd)
                accum[dd] = fmaf(wlds[dg * 40 + dd][c], xv, accum[dd]);
        }
        __syncthreads();
    }

    const int n = n0 + nl;
#pragma unroll
    for (int dd = 0; dd < 40; ++dd) {
        int d = dg * 40 + dd;
        f16 h = (f16)(accum[dd] + bcat[d]);
        if (d < 32)       Qb[((size_t)b * NPIX + n) * CQK + d]          = h;
        else if (d < 64)  Kb[((size_t)b * NPIX + n) * CQK + (d - 32)]   = h;
        else              Vb[((size_t)b * CIN + (d - 64)) * NPIX + n]   = h;
    }
}

// ---------------------------------------------------------------------------
// Kernel 3: flash attention, fully register-resident (no LDS).
// Scores computed transposed: S' = K x Q^T  -> D rows = keys, cols = queries.
//   => per-lane softmax stats are scalars (query = lane%16), one shfl_xor(16)
//      completes the cross-half reduction.
// P.V computed as out^T = V x P: A = V tile, B = P fragment. The transposed
// score tile is already in B-layout except 8 f16 pairs owned by the partner
// lane (lane^16): exchanged with 4 packed shfl_xor + cndmask selects.
// Block = 256 thr = 8 waves; wave w: query row-tile (w&1), channels (w>>1)*64.
// 64 keys per iteration: 4 score WMMAs + 8 PV WMMAs.
// ---------------------------------------------------------------------------
__global__ __launch_bounds__(256) void attn_kernel(
    const float* __restrict__ x,
    const f16* __restrict__ Qb, const f16* __restrict__ Kb,
    const f16* __restrict__ Vb, const float* __restrict__ gamma,
    float* __restrict__ out) {
    const int t    = threadIdx.x;
    const int w    = t >> 5;
    const int lane = t & 31;
    const int ln   = lane & 15;
    const int half = lane >> 4;
    const int b    = blockIdx.x >> 7;             // 128 query-blocks per batch
    const int qblk = blockIdx.x & 127;
    const int i0   = qblk * 32 + (w & 1) * 16;    // query tile base
    const int c0   = (w >> 1) * 64;               // channel group base

    // Q B-fragment (loop invariant): col n = ln, k = half*16 .. +15 contiguous
    const v16h qbf = *(const v16h*)(Qb + ((size_t)b * NPIX + i0 + ln) * CQK + half * 16);

    v8f vzero;
#pragma unroll
    for (int i = 0; i < 8; ++i) vzero[i] = 0.0f;
    v8f acc[4];
#pragma unroll
    for (int c = 0; c < 4; ++c) acc[c] = vzero;

    float rowmax = -3.0e38f;   // per-lane: stats for query (i0 + ln)
    float rowsum = 0.0f;

    const f16* kbase = Kb + (size_t)b * NPIX * CQK;
    const f16* vbase = Vb + ((size_t)b * CIN + c0 + ln) * NPIX;

    for (int j0 = 0; j0 < NPIX; j0 += 64) {
        // ---- 4 score tiles: S'[key m][query n], A = K tile (A-layout) ------
        v8f s[4];
#pragma unroll
        for (int kt = 0; kt < 4; ++kt) {
            const f16* krow = kbase + (size_t)(j0 + kt * 16 + ln) * CQK + half * 16;
            v8h lo = *(const v8h*)krow;
            v8h hi = *(const v8h*)(krow + 8);
            v16h ka;
#pragma unroll
            for (int i = 0; i < 8; ++i) { ka[i] = lo[i]; ka[i + 8] = hi[i]; }
            s[kt] = __builtin_amdgcn_wmma_f32_16x16x32_f16(false, ka, false, qbf,
                                                           (short)0, vzero, false, false);
        }
        if (j0 + 64 < NPIX) {  // prefetch next K block (global_prefetch_b8)
            __builtin_prefetch(kbase + (size_t)(j0 + 64 + lane) * CQK, 0, 0);
            __builtin_prefetch(kbase + (size_t)(j0 + 96 + lane) * CQK, 0, 0);
        }

        // ---- per-lane max over this lane's 32 keys, then cross-half --------
        float lm = s[0][0];
#pragma unroll
        for (int kt = 0; kt < 4; ++kt)
#pragma unroll
            for (int i = 0; i < 8; ++i) lm = fmaxf(lm, s[kt][i]);
        lm = fmaxf(lm, __shfl_xor(lm, 16, 32));

        float nm   = fmaxf(rowmax, lm);
        float corr = exp2f((rowmax - nm) * LOG2E);
        rowmax = nm;

        // ---- exponentials, packed f16x2 --------------------------------
        int ppk[4][4];
        float psum = 0.0f;
#pragma unroll
        for (int kt = 0; kt < 4; ++kt)
#pragma unroll
            for (int j = 0; j < 4; ++j) {
                float pa = exp2f((s[kt][2 * j]     - nm) * LOG2E);
                float pb = exp2f((s[kt][2 * j + 1] - nm) * LOG2E);
                psum += pa + pb;
                v2h ph; ph[0] = (f16)pa; ph[1] = (f16)pb;
                ppk[kt][j] = __builtin_bit_cast(int, ph);
            }
        rowsum = rowsum * corr + psum;
#pragma unroll
        for (int c = 0; c < 4; ++c)
#pragma unroll
            for (int i = 0; i < 8; ++i) acc[c][i] *= corr;

        // ---- assemble P B-fragments (kk pairs: tiles {0,1} and {2,3}) -----
        // half=0 owns kk kbase+0..7 (its tile), partner supplies kk kbase+8..15 etc.
        v16h pbf[2];
#pragma unroll
        for (int pp = 0; pp < 2; ++pp) {
            int lo32[4], hi32[4];
#pragma unroll
            for (int j = 0; j < 4; ++j) {
                int own0 = ppk[2 * pp][j];      // tile pp*2   (keys base+0..15)
                int own1 = ppk[2 * pp + 1][j];  // tile pp*2+1 (keys base+16..31)
                int send = half ? own0 : own1;  // what partner needs
                int keep = half ? own1 : own0;  // what we use locally
                int exch = __shfl_xor(send, 16, 32);
                lo32[j] = half ? exch : keep;
                hi32[j] = half ? keep : exch;
            }
            v8i tmp;
#pragma unroll
            for (int j = 0; j < 4; ++j) { tmp[j] = lo32[j]; tmp[4 + j] = hi32[j]; }
            pbf[pp] = __builtin_bit_cast(v16h, tmp);
        }

        // ---- out^T += V x P : A = V tile (A-layout), B = P ----------------
#pragma unroll
        for (int ct = 0; ct < 4; ++ct) {
#pragma unroll
            for (int pp = 0; pp < 2; ++pp) {
                const f16* vrow = vbase + (size_t)ct * 16 * NPIX + j0 + pp * 32 + half * 16;
                v8h lo = *(const v8h*)vrow;
                v8h hi = *(const v8h*)(vrow + 8);
                v16h va;
#pragma unroll
                for (int i = 0; i < 8; ++i) { va[i] = lo[i]; va[i + 8] = hi[i]; }
                acc[ct] = __builtin_amdgcn_wmma_f32_16x16x32_f16(false, va, false, pbf[pp],
                                                                 (short)0, acc[ct], false, false);
            }
        }
    }

    // ---- finalize: one cross-half add, lane-uniform 1/sum, gamma*out + x --
    rowsum += __shfl_xor(rowsum, 16, 32);
    const float inv = 1.0f / rowsum;
    const float g   = gamma[0];
#pragma unroll
    for (int ct = 0; ct < 4; ++ct) {
#pragma unroll
        for (int r = 0; r < 8; ++r) {
            int ch = c0 + ct * 16 + r + half * 8;      // D row = channel
            size_t idx = ((size_t)b * CIN + ch) * NPIX + i0 + ln;  // D col = query
            out[idx] = g * (acc[ct][r] * inv) + x[idx];
        }
    }
}

// ---------------------------------------------------------------------------
extern "C" void kernel_launch(void* const* d_in, const int* in_sizes, int n_in,
                              void* d_out, int out_size, void* d_ws, size_t ws_size,
                              hipStream_t stream) {
    const float* x     = (const float*)d_in[0];
    const float* wq    = (const float*)d_in[1];
    const float* bq    = (const float*)d_in[2];
    const float* wk    = (const float*)d_in[3];
    const float* bk    = (const float*)d_in[4];
    const float* wv    = (const float*)d_in[5];
    const float* bv    = (const float*)d_in[6];
    const float* gamma = (const float*)d_in[7];
    float* out = (float*)d_out;

    char* ws = (char*)d_ws;
    float* Wcat = (float*)(ws + 0);          // 320*256*4 = 327680 B
    float* bcat = (float*)(ws + 327680);     // 320*4     (pad to 4096)
    f16*   Qb   = (f16*)(ws + 331776);       // 4*4096*32*2 = 1 MiB
    f16*   Kb   = (f16*)(ws + 1380352);      // 1 MiB
    f16*   Vb   = (f16*)(ws + 2428928);      // 4*256*4096*2 = 8 MiB (end ~10.3 MiB)

    concat_w_kernel<<<320, 256, 0, stream>>>(wq, bq, wk, bk, wv, bv, Wcat, bcat);
    proj_kernel<<<NB * (NPIX / 64), 512, 0, stream>>>(x, Wcat, bcat, Qb, Kb, Vb);
    attn_kernel<<<NB * (NPIX / 32), 256, 0, stream>>>(x, Qb, Kb, Vb, gamma, out);
}
